// Seq2SeqModel_30099130810582
// MI455X (gfx1250) — matrix operand order repaired and checked by reading, not compiled, not run on hardware
//
#include <hip/hip_runtime.h>

typedef unsigned short ushort_t;
typedef __attribute__((ext_vector_type(16))) __bf16 v16bf;
typedef __attribute__((ext_vector_type(8)))  __bf16 v8bf;
typedef __attribute__((ext_vector_type(8)))  float  v8f;

#if defined(__HIP_DEVICE_COMPILE__) && \
    __has_builtin(__builtin_amdgcn_global_load_async_to_lds_b128) && \
    __has_builtin(__builtin_amdgcn_s_wait_asynccnt)
#define USE_ASYNC_STAGE 1
#else
#define USE_ASYNC_STAGE 0
#endif

#define GAS __attribute__((address_space(1)))
#define LAS __attribute__((address_space(3)))

// exact pointee type of the async-copy builtin params: int4 (gcc vector_size)
typedef int v4i_gcc __attribute__((__vector_size__(16)));
typedef GAS v4i_gcc* gv4i_p;
typedef LAS v4i_gcc* lv4i_p;

// ---------- helpers ----------
__device__ __forceinline__ ushort_t f32_to_bf16(float f) {
    unsigned u = __float_as_uint(f);
    unsigned r = (u + 0x7FFFu + ((u >> 16) & 1u)) >> 16;   // RNE
    return (ushort_t)r;
}

// A-matrix fragment (16-bit 16x32, ISA layout): lane half h:
//   elems 0..7  <- K = k0 + 8*h .. +7      (VGPRs 0..3)
//   elems 8..15 <- K = k0 + 16 + 8*h .. +7 (VGPRs 4..7)
__device__ __forceinline__ v16bf load_frag_a(const ushort_t* row, int k0, int half) {
    const v8bf lo = *(const v8bf*)(row + k0 + 8 * half);
    const v8bf hi = *(const v8bf*)(row + k0 + 16 + 8 * half);
    return __builtin_shufflevector(lo, hi, 0,1,2,3,4,5,6,7,8,9,10,11,12,13,14,15);
}

// B-matrix fragment (16-bit 32x16, ISA layout): lane half h holds
// K = k0 + 8*h .. +7 then K = k0 + 16 + 8*h .. +7 (column n = lane%16).
__device__ __forceinline__ v16bf load_frag_b(const ushort_t* row, int k0, int half) {
    const v8bf lo = *(const v8bf*)(row + k0 + 8 * half);
    const v8bf hi = *(const v8bf*)(row + k0 + 16 + 8 * half);
    return __builtin_shufflevector(lo, hi, 0,1,2,3,4,5,6,7,8,9,10,11,12,13,14,15);
}

// ---------- elementwise f32 -> bf16 ----------
__global__ void s2s_cvt_bf16(const float* __restrict__ src, ushort_t* __restrict__ dst, int n) {
    int i = blockIdx.x * blockDim.x + threadIdx.x;
    if (i < n) dst[i] = f32_to_bf16(src[i]);
}

// ---------- embedding gather -> bf16 rows [nRows,256] ----------
__global__ void s2s_gather(const float* __restrict__ emb, const int* __restrict__ tok,
                           ushort_t* __restrict__ dst, int T, int tokStride, int nRows) {
    int idx = blockIdx.x * blockDim.x + threadIdx.x;
    int m = idx >> 8, h = idx & 255;
    if (m < nRows) {
        int b = m / T, t = m % T;
        int tk = tok[b * tokStride + t];
        dst[idx] = f32_to_bf16(emb[(size_t)tk * 256 + h]);
    }
}

// ---------- LDS-staged double-buffered WMMA GEMM ----------
// C[M,N] = A_bf16[M,K] @ B_bf16[N,K]^T + bias.  block=256 (8 waves),
// macro tile 128x128, wave tile 32(M)x64(N), K-slab 32.
// Staging uses CDNA5 async global->LDS copies (ASYNCcnt) when available.
// grid = (N/128, ceil(M/128)); A buffer must have >= gridDim.y*128 rows.
#define TK 32
#define LDT 40   // padded LDS row stride (bf16 elems); 80B -> conflict-free frags

__global__ __launch_bounds__(256)
void s2s_wmma_gemm(const ushort_t* __restrict__ A, const ushort_t* __restrict__ Bm,
                   const float* __restrict__ bias, float* __restrict__ C,
                   int M, int N, int K, int ntC) {
    __shared__ __align__(16) ushort_t As[2][128 * LDT];
    __shared__ __align__(16) ushort_t Bs[2][128 * LDT];

    const int tid  = threadIdx.x;
    const int lane = tid & 31;
    const int wave = tid >> 5;
    const int half = lane >> 4;
    const int r    = lane & 15;
    const int wm   = (wave & 3) * 32;     // wave M offset inside macro tile
    const int wn   = (wave >> 2) * 64;    // wave N offset inside macro tile
    const int m0   = blockIdx.y * 128;
    const int n0   = blockIdx.x * 128;

    // staging: transfer q = tid + 256*j -> row = q>>2, seg = (q&3)*8 elems
    const int row0 = tid >> 2,          seg0 = (tid & 3) * 8;
    const int row1 = (tid + 256) >> 2,  seg1 = ((tid + 256) & 3) * 8;

    const ushort_t* Ab = A  + (size_t)m0 * K;
    const ushort_t* Bb = Bm + (size_t)n0 * K;

    v8f acc[8] = {v8f{}, v8f{}, v8f{}, v8f{}, v8f{}, v8f{}, v8f{}, v8f{}};
    const int nK = K / TK;

#if USE_ASYNC_STAGE
    // ---- CDNA5 async global->LDS staging (no VGPR round trip) ----
    auto stage = [&](int buf, int k0) {
        __builtin_amdgcn_global_load_async_to_lds_b128(
            (gv4i_p)(Ab + (size_t)row0 * K + k0 + seg0),
            (lv4i_p)&As[buf][row0 * LDT + seg0], 0, 0);
        __builtin_amdgcn_global_load_async_to_lds_b128(
            (gv4i_p)(Ab + (size_t)row1 * K + k0 + seg1),
            (lv4i_p)&As[buf][row1 * LDT + seg1], 0, 0);
        __builtin_amdgcn_global_load_async_to_lds_b128(
            (gv4i_p)(Bb + (size_t)row0 * K + k0 + seg0),
            (lv4i_p)&Bs[buf][row0 * LDT + seg0], 0, 0);
        __builtin_amdgcn_global_load_async_to_lds_b128(
            (gv4i_p)(Bb + (size_t)row1 * K + k0 + seg1),
            (lv4i_p)&Bs[buf][row1 * LDT + seg1], 0, 0);
    };
    stage(0, 0);
    for (int ks = 0; ks < nK; ++ks) {
        const int  cur = ks & 1;
        const bool pf  = (ks + 1) < nK;
        if (pf) {   // slab ks+1 in flight while slab ks computes
            stage(cur ^ 1, (ks + 1) * TK);
            __builtin_amdgcn_s_wait_asynccnt(4);   // in-order: slab ks complete
        } else {
            __builtin_amdgcn_s_wait_asynccnt(0);
        }
        __syncthreads();
        const ushort_t* Ap = &As[cur][0];
        const ushort_t* Bp = &Bs[cur][0];
        v16bf a0 = load_frag_a(Ap + (wm + r) * LDT, 0, half);
        v16bf a1 = load_frag_a(Ap + (wm + 16 + r) * LDT, 0, half);
#pragma unroll
        for (int i = 0; i < 4; ++i) {
            v16bf b = load_frag_b(Bp + (wn + 16 * i + r) * LDT, 0, half);
            acc[i]     = __builtin_amdgcn_wmma_f32_16x16x32_bf16(
                             false, a0, false, b, (short)0, acc[i],     false, false);
            acc[4 + i] = __builtin_amdgcn_wmma_f32_16x16x32_bf16(
                             false, a1, false, b, (short)0, acc[4 + i], false, false);
        }
        __syncthreads();   // all waves done with buf cur before it is restaged
    }
#else
    // ---- fallback: global->VGPR->LDS double buffer ----
    uint4 ar0 = *(const uint4*)(Ab + (size_t)row0 * K + seg0);
    uint4 ar1 = *(const uint4*)(Ab + (size_t)row1 * K + seg1);
    uint4 br0 = *(const uint4*)(Bb + (size_t)row0 * K + seg0);
    uint4 br1 = *(const uint4*)(Bb + (size_t)row1 * K + seg1);
    *(uint4*)(&As[0][row0 * LDT + seg0]) = ar0;
    *(uint4*)(&As[0][row1 * LDT + seg1]) = ar1;
    *(uint4*)(&Bs[0][row0 * LDT + seg0]) = br0;
    *(uint4*)(&Bs[0][row1 * LDT + seg1]) = br1;
    __syncthreads();

    for (int ks = 0; ks < nK; ++ks) {
        const int  cur = ks & 1;
        const bool pf  = (ks + 1) < nK;
        if (pf) {
            const int k0 = (ks + 1) * TK;
            ar0 = *(const uint4*)(Ab + (size_t)row0 * K + k0 + seg0);
            ar1 = *(const uint4*)(Ab + (size_t)row1 * K + k0 + seg1);
            br0 = *(const uint4*)(Bb + (size_t)row0 * K + k0 + seg0);
            br1 = *(const uint4*)(Bb + (size_t)row1 * K + k0 + seg1);
        }
        const ushort_t* Ap = &As[cur][0];
        const ushort_t* Bp = &Bs[cur][0];
        v16bf a0 = load_frag_a(Ap + (wm + r) * LDT, 0, half);
        v16bf a1 = load_frag_a(Ap + (wm + 16 + r) * LDT, 0, half);
#pragma unroll
        for (int i = 0; i < 4; ++i) {
            v16bf b = load_frag_b(Bp + (wn + 16 * i + r) * LDT, 0, half);
            acc[i]     = __builtin_amdgcn_wmma_f32_16x16x32_bf16(
                             false, a0, false, b, (short)0, acc[i],     false, false);
            acc[4 + i] = __builtin_amdgcn_wmma_f32_16x16x32_bf16(
                             false, a1, false, b, (short)0, acc[4 + i], false, false);
        }
        if (pf) {
            const int nxt = cur ^ 1;
            *(uint4*)(&As[nxt][row0 * LDT + seg0]) = ar0;
            *(uint4*)(&As[nxt][row1 * LDT + seg1]) = ar1;
            *(uint4*)(&Bs[nxt][row0 * LDT + seg0]) = br0;
            *(uint4*)(&Bs[nxt][row1 * LDT + seg1]) = br1;
        }
        __syncthreads();
    }
#endif

#pragma unroll
    for (int sub = 0; sub < 2; ++sub) {
#pragma unroll
        for (int i = 0; i < 4; ++i) {
            int ncol = n0 + wn + 16 * i + r;
            float bv = bias ? bias[ncol] : 0.0f;
#pragma unroll
            for (int v = 0; v < 8; ++v) {
                int mrow = m0 + wm + sub * 16 + v + 8 * half;
                if (mrow < M) {
                    float val = acc[sub * 4 + i][v] + bv;
                    float* p = &C[(size_t)mrow * N + ncol];
                    if (ntC) __builtin_nontemporal_store(val, p);  // TH=NT: don't evict fc_W from L2
                    else     *p = val;
                }
            }
        }
    }
}

// ---------- GRU recurrence ----------
// One block = 16 batch rows for the whole T loop. h resident in LDS (f32 + bf16).
// Per step: gh[16,768] = h @ Whh^T via WMMA (16 waves x 3 N-tiles x 8 K-steps,
// software-pipelined B loads), then fused gates.
__global__ __launch_bounds__(512)
void s2s_gru(const ushort_t* __restrict__ Whh, const float* __restrict__ bhh,
             const float* __restrict__ xp, const float* __restrict__ h0,
             float* __restrict__ out, float* __restrict__ hT, int T) {
    __shared__ __align__(16) float    h_f32[16 * 256];
    __shared__ __align__(16) ushort_t h_b16[16 * 256];
    __shared__ __align__(16) float    gh[16 * 768];

    const int b0   = blockIdx.x * 16;
    const int lane = threadIdx.x & 31;
    const int wave = threadIdx.x >> 5;
    const int half = lane >> 4;
    const int r    = lane & 15;

    for (int e = threadIdx.x; e < 16 * 256; e += 512) {
        float v = h0 ? h0[(size_t)(b0 + (e >> 8)) * 256 + (e & 255)] : 0.0f;
        h_f32[e] = v;
        h_b16[e] = f32_to_bf16(v);
    }
    __syncthreads();

    const ushort_t* hrow = h_b16 + r * 256;
    int nc[3];
    const ushort_t* brow[3];
#pragma unroll
    for (int i = 0; i < 3; ++i) {
        nc[i]   = (wave * 3 + i) * 16;
        brow[i] = Whh + (size_t)(nc[i] + r) * 256;
    }

    for (int t = 0; t < T; ++t) {
        v8f acc[3] = {v8f{}, v8f{}, v8f{}};
        v16bf af = load_frag_a(hrow, 0, half);
        v16bf bc[3];
#pragma unroll
        for (int i = 0; i < 3; ++i) bc[i] = load_frag_b(brow[i], 0, half);

#pragma unroll
        for (int kk = 0; kk < 8; ++kk) {
            v16bf an;
            v16bf bn[3];
            if (kk < 7) {   // prefetch next K-slab fragments
                an = load_frag_a(hrow, (kk + 1) * 32, half);
#pragma unroll
                for (int i = 0; i < 3; ++i) bn[i] = load_frag_b(brow[i], (kk + 1) * 32, half);
            }
#pragma unroll
            for (int i = 0; i < 3; ++i)
                acc[i] = __builtin_amdgcn_wmma_f32_16x16x32_bf16(
                             false, af, false, bc[i], (short)0, acc[i], false, false);
            if (kk < 7) {
                af = an;
#pragma unroll
                for (int i = 0; i < 3; ++i) bc[i] = bn[i];
            }
        }
#pragma unroll
        for (int i = 0; i < 3; ++i)
#pragma unroll
            for (int v = 0; v < 8; ++v)
                gh[(v + 8 * half) * 768 + nc[i] + r] = acc[i][v];
        __syncthreads();

        for (int e = threadIdx.x; e < 16 * 256; e += 512) {
            int rb = e >> 8, cc = e & 255;
            size_t xrow = ((size_t)(b0 + rb) * T + t) * 768;
            float ghr = gh[rb * 768 + cc]       + bhh[cc];
            float ghz = gh[rb * 768 + 256 + cc] + bhh[256 + cc];
            float ghn = gh[rb * 768 + 512 + cc] + bhh[512 + cc];
            float rg = 1.0f / (1.0f + __expf(-(xp[xrow + cc] + ghr)));
            float zg = 1.0f / (1.0f + __expf(-(xp[xrow + 256 + cc] + ghz)));
            float ng = tanhf(xp[xrow + 512 + cc] + rg * ghn);
            float hn = (1.0f - zg) * ng + zg * h_f32[e];
            h_f32[e] = hn;
            h_b16[e] = f32_to_bf16(hn);
            out[((size_t)(b0 + rb) * T + t) * 256 + cc] = hn;
        }
        __syncthreads();
    }

    if (hT) {
        for (int e = threadIdx.x; e < 16 * 256; e += 512)
            hT[(size_t)(b0 + (e >> 8)) * 256 + (e & 255)] = h_f32[e];
    }
}

// ---------- attention: one block per batch element ----------
// Stages enc_out[b] (128x256 f32, padded stride 260) into dynamic LDS once,
// then loops over all 127 decoder steps: score -> mask -> softmax -> ctx,
// emits feats rows ([0:256)=dec_out, [256:512)=ctx) in bf16.
#define EPAD 260

__global__ __launch_bounds__(256)
void s2s_attn(const float* __restrict__ enc_out, const float* __restrict__ dec_out,
              const int* __restrict__ x, ushort_t* __restrict__ feats) {
    extern __shared__ __align__(16) char smem_raw[];
    float* enc_s = (float*)smem_raw;            // 128 * EPAD
    float* dvec  = enc_s + 128 * EPAD;          // 256
    float* wsc   = dvec + 256;                  // 128
    float* part  = wsc + 128;                   // 256
    int*   msk   = (int*)(part + 256);          // 128

    const int b   = blockIdx.x;
    const int tid = threadIdx.x;

    for (int idx = tid; idx < 128 * 256; idx += 256)
        enc_s[(idx >> 8) * EPAD + (idx & 255)] = enc_out[(size_t)b * 128 * 256 + idx];
    if (tid < 128) msk[tid] = x[b * 128 + tid];
    __syncthreads();

    const int s  = tid & 127;
    const int hh = tid >> 7;

    for (int t = 0; t < 127; ++t) {
        const int m = b * 127 + t;
        dvec[tid] = dec_out[(size_t)m * 256 + tid];
        __syncthreads();

        // scores: two threads per source position, 128-length partial dots
        const float* er = enc_s + s * EPAD + hh * 128;
        const float* dp = dvec + hh * 128;
        float a = 0.0f;
        for (int h = 0; h < 128; ++h) a += dp[h] * er[h];
        part[tid] = a;
        __syncthreads();
        if (tid < 128) {
            float sc = part[tid] + part[tid + 128];
            wsc[tid] = (msk[tid] == 1) ? sc : -10000.0f;
        }
        __syncthreads();

        // softmax over 128
        if (tid < 128) part[tid] = wsc[tid];
        __syncthreads();
        for (int o = 64; o > 0; o >>= 1) {
            if (tid < o) part[tid] = fmaxf(part[tid], part[tid + o]);
            __syncthreads();
        }
        float mx = part[0];
        __syncthreads();
        if (tid < 128) {
            float e = __expf(wsc[tid] - mx);
            wsc[tid] = e;
            part[tid] = e;
        }
        __syncthreads();
        for (int o = 64; o > 0; o >>= 1) {
            if (tid < o) part[tid] += part[tid + o];
            __syncthreads();
        }
        float inv = 1.0f / part[0];
        __syncthreads();

        // ctx over h = tid (LDS reads conflict-free: consecutive tid)
        float c = 0.0f;
        for (int ss = 0; ss < 128; ++ss) c += wsc[ss] * enc_s[ss * EPAD + tid];
        c *= inv;

        ushort_t* frow = feats + (size_t)m * 512;
        frow[tid]       = f32_to_bf16(dvec[tid]);
        frow[256 + tid] = f32_to_bf16(c);
        __syncthreads();
    }
}

// ---------- host ----------
extern "C" void kernel_launch(void* const* d_in, const int* in_sizes, int n_in,
                              void* d_out, int out_size, void* d_ws, size_t ws_size,
                              hipStream_t stream) {
    (void)in_sizes; (void)n_in; (void)out_size; (void)ws_size;
    const int* x         = (const int*)d_in[0];
    const int* y         = (const int*)d_in[1];
    const float* enc_emb = (const float*)d_in[2];
    const float* dec_emb = (const float*)d_in[3];
    const float* enc_Wih = (const float*)d_in[4];
    const float* enc_Whh = (const float*)d_in[5];
    const float* enc_bih = (const float*)d_in[6];
    const float* enc_bhh = (const float*)d_in[7];
    const float* dec_Wih = (const float*)d_in[8];
    const float* dec_Whh = (const float*)d_in[9];
    const float* dec_bih = (const float*)d_in[10];
    const float* dec_bhh = (const float*)d_in[11];
    const float* fc_W    = (const float*)d_in[12];
    const float* fc_b    = (const float*)d_in[13];
    float* out = (float*)d_out;

    char* w = (char*)d_ws;
    size_t off = 0;
    auto alloc = [&](size_t bytes) -> void* {
        void* p = w + off;
        off = (off + bytes + 255) & ~(size_t)255;
        return p;
    };
    // A-side bf16 buffers padded to 4096 rows (GEMM macro tile = 128 rows)
    ushort_t* fcW_bf   = (ushort_t*)alloc((size_t)32000 * 512 * 2);
    ushort_t* eWih_bf  = (ushort_t*)alloc((size_t)768 * 256 * 2);
    ushort_t* eWhh_bf  = (ushort_t*)alloc((size_t)768 * 256 * 2);
    ushort_t* dWih_bf  = (ushort_t*)alloc((size_t)768 * 256 * 2);
    ushort_t* dWhh_bf  = (ushort_t*)alloc((size_t)768 * 256 * 2);
    ushort_t* xe_bf    = (ushort_t*)alloc((size_t)4096 * 256 * 2);
    ushort_t* yd_bf    = (ushort_t*)alloc((size_t)4096 * 256 * 2);
    float*    xp_enc   = (float*)alloc((size_t)4096 * 768 * 4);
    float*    xp_dec   = (float*)alloc((size_t)4064 * 768 * 4);
    float*    enc_outF = (float*)alloc((size_t)4096 * 256 * 4);
    float*    dec_outF = (float*)alloc((size_t)4064 * 256 * 4);
    float*    h_fin    = (float*)alloc((size_t)32 * 256 * 4);
    ushort_t* feats_bf = (ushort_t*)alloc((size_t)4096 * 512 * 2);

    const int nW = 768 * 256;
    s2s_cvt_bf16<<<(nW + 255) / 256, 256, 0, stream>>>(enc_Wih, eWih_bf, nW);
    s2s_cvt_bf16<<<(nW + 255) / 256, 256, 0, stream>>>(enc_Whh, eWhh_bf, nW);
    s2s_cvt_bf16<<<(nW + 255) / 256, 256, 0, stream>>>(dec_Wih, dWih_bf, nW);
    s2s_cvt_bf16<<<(nW + 255) / 256, 256, 0, stream>>>(dec_Whh, dWhh_bf, nW);
    const int nFc = 32000 * 512;
    s2s_cvt_bf16<<<(nFc + 255) / 256, 256, 0, stream>>>(fc_W, fcW_bf, nFc);

    s2s_gather<<<4096, 256, 0, stream>>>(enc_emb, x, xe_bf, 128, 128, 4096);
    s2s_gather<<<4064, 256, 0, stream>>>(dec_emb, y, yd_bf, 127, 128, 4064);

    // xp = emb @ Wih^T + bih   (N=768, K=256)
    s2s_wmma_gemm<<<dim3(6, 32), 256, 0, stream>>>(xe_bf, eWih_bf, enc_bih, xp_enc, 4096, 768, 256, 0);
    s2s_wmma_gemm<<<dim3(6, 32), 256, 0, stream>>>(yd_bf, dWih_bf, dec_bih, xp_dec, 4064, 768, 256, 0);

    // recurrences (batch split across 2 blocks of 16 rows)
    s2s_gru<<<2, 512, 0, stream>>>(eWhh_bf, enc_bhh, xp_enc, nullptr, enc_outF, h_fin, 128);
    s2s_gru<<<2, 512, 0, stream>>>(dWhh_bf, dec_bhh, xp_dec, h_fin, dec_outF, nullptr, 127);

    // attention + feats packing (enc_out[b] staged in 136KB dynamic LDS)
    const int attn_smem = (128 * EPAD + 256 + 128 + 256) * 4 + 128 * 4;
    s2s_attn<<<32, 256, attn_smem, stream>>>(enc_outF, dec_outF, x, feats_bf);

    // logits = feats @ fc_W^T + fc_b  (dominant GEMM: 4064 x 32000 x 512, NT stores)
    s2s_wmma_gemm<<<dim3(250, 32), 256, 0, stream>>>(feats_bf, fcW_bf, fc_b, out, 4064, 32000, 512, 1);
}
